// MultiWindowCrossAttention_26749056319775
// MI455X (gfx1250) — matrix operand order repaired and checked by reading, not compiled, not run on hardware
//
#include <hip/hip_runtime.h>
#include <math.h>

typedef __bf16 bf16;
typedef __attribute__((ext_vector_type(16))) __bf16 v16bf;
typedef __attribute__((ext_vector_type(8)))  __bf16 v8bf;
typedef __attribute__((ext_vector_type(8)))  float   v8f;

#define CD    768
#define HDS   12
#define HD    64
#define LP    80            // L=65 padded to 5x16
#define NWIN  1024          // B(4) * 16 * 16 windows
#define MTOT  (NWIN * LP)   // 81920 rows
#define IMG   126
#define NTOK  (IMG * IMG)

// ---- WMMA helpers (CDNA5 bf16 16x16x32, f32 accum) -------------------------

static __device__ __forceinline__ v16bf frag16(const bf16* p0, const bf16* p1) {
  // two contiguous 8-half (16B) runs -> ds_load_b128 x2
  v8bf lo = *(const v8bf*)p0;
  v8bf hi = *(const v8bf*)p1;
  v16bf r;
#pragma unroll
  for (int i = 0; i < 8; ++i) { r[i] = lo[i]; r[i + 8] = hi[i]; }
  return r;
}

static __device__ __forceinline__ v8f wmma_bf16(v16bf a, v16bf b, v8f c) {
  return __builtin_amdgcn_wmma_f32_16x16x32_bf16(false, a, false, b, (short)0, c,
                                                 false, false);
}

// ---- weight convert f32 -> bf16 --------------------------------------------

__global__ void f32_to_bf16_kernel(const float* __restrict__ s,
                                   bf16* __restrict__ d, int n) {
  int i = blockIdx.x * blockDim.x + threadIdx.x;
  if (i < n) d[i] = (bf16)s[i];
}

// ---- window partition + qc append, f32 -> bf16 ------------------------------
// One block per padded window row (MTOT rows); rows 65..79 zero.

__global__ __launch_bounds__(256) void pack_kernel(const float* __restrict__ x,
                                                   const float* __restrict__ y,
                                                   const float* __restrict__ qc,
                                                   bf16* __restrict__ xw,
                                                   bf16* __restrict__ yw) {
  int row = blockIdx.x;                 // 0..MTOT-1
  int win = row / LP, t = row - win * LP;
  int b = win >> 8, widx = win & 255, wh = widx >> 4, ww = widx & 15;

  const float* sx = nullptr;
  const float* sy = nullptr;
  bool zero = false;
  if (t < 64) {
    int h = wh * 8 + (t >> 3) - 1;
    int w = ww * 8 + (t & 7) - 1;
    if (h >= 0 && h < IMG && w >= 0 && w < IMG) {
      size_t o = ((size_t)b * NTOK + (size_t)h * IMG + w) * CD;
      sx = x + o; sy = y + o;
    } else zero = true;
  } else if (t == 64) {
    sx = qc + (size_t)b * CD; sy = sx;
  } else zero = true;

  bf16* dx = xw + (size_t)row * CD;
  bf16* dy = yw + (size_t)row * CD;
  for (int c = threadIdx.x; c < CD; c += 256) {
    dx[c] = zero ? (bf16)0.f : (bf16)sx[c];
    dy[c] = zero ? (bf16)0.f : (bf16)sy[c];
  }
}

// ---- tiled WMMA GEMM: D = A[MTOT x 768] * W[768 x 768] + bias ---------------
// block tile 128x128, 8 waves of 32x64, K-step 32.
// mode 0: dst_bf[row*768+col] = bf16((acc+bias)*scale)
// mode 1: un-window + crop scatter of f32 (acc+bias) into d_out

__global__ __launch_bounds__(256) void wmma_gemm(const bf16* __restrict__ A,
                                                 const bf16* __restrict__ W,
                                                 const float* __restrict__ bias,
                                                 float scale, int mode,
                                                 bf16* __restrict__ dst_bf,
                                                 float* __restrict__ dst_f) {
  __shared__ __align__(16) bf16 As[128 * 40];   // [m][k] pitch 40 halfs
  __shared__ __align__(16) bf16 Bs[128 * 40];   // transposed: [n][k] pitch 40

  const int bm = blockIdx.x * 128, bn = blockIdx.y * 128;
  const int tid = threadIdx.x, wave = tid >> 5, lane = tid & 31;
  const int wm = (wave >> 1) * 32, wn = (wave & 1) * 64;
  const int g = lane >> 4, nn = lane & 15;

  v8f acc[2][4];
#pragma unroll
  for (int i = 0; i < 2; ++i)
#pragma unroll
    for (int j = 0; j < 4; ++j) acc[i][j] = (v8f){};

  const int arow = tid >> 1, ac = (tid & 1) * 16;   // 128 rows x 2 chunks
  const int brow = tid >> 3, bc = (tid & 7) * 16;   // 32 k-rows x 8 chunks
  const bf16* ap = A + (size_t)(bm + arow) * CD + ac;
  const bf16* wp = W + (size_t)brow * CD + bn + bc;

  for (int k0 = 0; k0 < CD; k0 += 32) {
    v8bf a0 = *(const v8bf*)(ap + k0);
    v8bf a1 = *(const v8bf*)(ap + k0 + 8);
    v8bf b0 = *(const v8bf*)(wp + (size_t)k0 * CD);
    v8bf b1 = *(const v8bf*)(wp + (size_t)k0 * CD + 8);
    if (k0 + 32 < CD) {
      __builtin_prefetch(ap + k0 + 32, 0, 1);
      __builtin_prefetch(wp + (size_t)(k0 + 32) * CD, 0, 1);
    }
    __syncthreads();
    *(v8bf*)(As + arow * 40 + ac) = a0;
    *(v8bf*)(As + arow * 40 + ac + 8) = a1;
#pragma unroll
    for (int e = 0; e < 8; ++e) {           // transpose-store B tile
      Bs[(bc + e) * 40 + brow]     = b0[e];
      Bs[(bc + 8 + e) * 40 + brow] = b1[e];
    }
    __syncthreads();

    v16bf af[2], bg[4];
#pragma unroll
    for (int i = 0; i < 2; ++i) {
      const bf16* p = As + (wm + 16 * i + nn) * 40;   // m = lane&15
      af[i] = frag16(p + 16 * g, p + 16 * g + 8);     // K layout: 16g + e
    }
#pragma unroll
    for (int j = 0; j < 4; ++j) {
      const bf16* p = Bs + (wn + 16 * j + nn) * 40;   // n = lane&15
      bg[j] = frag16(p + 16 * g, p + 16 * g + 8);     // K layout: 16g + e
    }
#pragma unroll
    for (int i = 0; i < 2; ++i)
#pragma unroll
      for (int j = 0; j < 4; ++j) acc[i][j] = wmma_bf16(af[i], bg[j], acc[i][j]);
  }

#pragma unroll
  for (int i = 0; i < 2; ++i)
#pragma unroll
    for (int j = 0; j < 4; ++j) {
      int col = bn + wn + 16 * j + nn;
      float bvl = bias[col];
#pragma unroll
      for (int r = 0; r < 8; ++r) {
        int row = bm + wm + 16 * i + r + 8 * g;
        float v = (acc[i][j][r] + bvl) * scale;
        if (mode == 0) {
          dst_bf[(size_t)row * CD + col] = (bf16)v;
        } else {
          int win = row / LP, t = row - win * LP;
          if (t < 64) {
            int b = win >> 8, widx = win & 255, wh = widx >> 4, ww = widx & 15;
            int h = wh * 8 + (t >> 3) - 1, w = ww * 8 + (t & 7) - 1;
            if (h >= 0 && h < IMG && w >= 0 && w < IMG)
              dst_f[((size_t)b * NTOK + (size_t)h * IMG + w) * CD + col] = v;
          }
        }
      }
    }
}

// ---- fused window attention: one block (128 thr / 4 waves) per (win, head) --

__global__ __launch_bounds__(128) void attn_kernel(const bf16* __restrict__ Q,
                                                   const bf16* __restrict__ K,
                                                   const bf16* __restrict__ V,
                                                   const float* __restrict__ rpe,
                                                   bf16* __restrict__ O) {
  __shared__ __align__(16) char smem[64528];
  bf16*  Qs   = (bf16*)smem;                 // 80x72 halfs  (11520 B)
  bf16*  Ks   = (bf16*)(smem + 11520);       // 80x72 halfs  (-> 23040)
  bf16*  Ps   = (bf16*)smem;                 // 80x104 halfs, ALIASES Qs/Ks
  bf16*  Vt   = (bf16*)(smem + 23040);       // 64x104 halfs (-> 36352), [d][key]
  float* Ss   = (float*)(smem + 36352);      // 80x84 f32    (-> 63232)
  float* rpeL = (float*)(smem + 63232);      // 228 f32
  float* mk   = (float*)(smem + 64144);      // 96 f32 key mask

  const int win = blockIdx.x, head = blockIdx.y;
  const int tid = threadIdx.x, wave = tid >> 5, lane = tid & 31;
  const int g = lane >> 4, nn = lane & 15;
  const int widx = win & 255, wh = widx >> 4, ww = widx & 15;

  const bf16* qb = Q + ((size_t)win * LP) * CD + head * HD;
  const bf16* kb = K + ((size_t)win * LP) * CD + head * HD;
  const bf16* vb = V + ((size_t)win * LP) * CD + head * HD;

  for (int c = tid; c < 640; c += 128) {          // 80 rows x 8 chunks of 8
    int row = c >> 3, d0 = (c & 7) * 8;
    *(v8bf*)(Qs + row * 72 + d0) = *(const v8bf*)(qb + (size_t)row * CD + d0);
    *(v8bf*)(Ks + row * 72 + d0) = *(const v8bf*)(kb + (size_t)row * CD + d0);
    v8bf vv = *(const v8bf*)(vb + (size_t)row * CD + d0);
#pragma unroll
    for (int e = 0; e < 8; ++e) Vt[(d0 + e) * 104 + row] = vv[e];
  }
  for (int c = tid; c < 1024; c += 128)           // zero Vt key cols 80..95
    Vt[(c >> 4) * 104 + 80 + (c & 15)] = (bf16)0.f;
  for (int c = tid; c < 228; c += 128) rpeL[c] = rpe[c * HDS + head];
  if (tid < 96) {
    float m = -INFINITY;
    if (tid < 64) {
      int h = wh * 8 + (tid >> 3) - 1, w = ww * 8 + (tid & 7) - 1;
      if (h >= 0 && h < IMG && w >= 0 && w < IMG) m = 0.f;
    } else if (tid == 64) m = 0.f;                // qc key never masked
    mk[tid] = m;
  }
  __syncthreads();

  // S = Q * K^T : 5x5 tiles, K-dim 64 = 2 steps
  for (int t = wave; t < 25; t += 4) {
    int rt = t / 5, ct = t % 5;
    v8f s = (v8f){};
#pragma unroll
    for (int ks = 0; ks < 2; ++ks) {
      const bf16* pa = Qs + (rt * 16 + nn) * 72 + ks * 32;
      const bf16* pb = Ks + (ct * 16 + nn) * 72 + ks * 32;
      v16bf a = frag16(pa + 16 * g, pa + 16 * g + 8);
      v16bf b = frag16(pb + 16 * g, pb + 16 * g + 8);
      s = wmma_bf16(a, b, s);
    }
#pragma unroll
    for (int r = 0; r < 8; ++r) {
      int qi = rt * 16 + r + 8 * g, kj = ct * 16 + nn;
      int idx;
      if (qi >= 64)      idx = (kj >= 64) ? 227 : 225;
      else if (kj >= 64) idx = 226;
      else idx = ((qi >> 3) - (kj >> 3) + 7) * 15 + ((qi & 7) - (kj & 7) + 7);
      Ss[qi * 84 + kj] = s[r] + rpeL[idx] + mk[kj];
    }
  }
  __syncthreads();

  // zero P (Qs/Ks are dead now; Ps aliases them)
  for (int c = tid; c < LP * 104; c += 128) Ps[c] = (bf16)0.f;
  __syncthreads();

  // softmax rows 0..64 over 80 cols (cols >=65 are -inf)
  if (tid < 65) {
    float* srow = Ss + tid * 84;
    float mx = -INFINITY;
    for (int j = 0; j < 80; ++j) mx = fmaxf(mx, srow[j]);
    float sum = 0.f;
    for (int j = 0; j < 80; ++j) { float e = __expf(srow[j] - mx); srow[j] = e; sum += e; }
    float inv = 1.f / sum;
    bf16* prow = Ps + tid * 104;
    for (int j = 0; j < 80; ++j) prow[j] = (bf16)(srow[j] * inv);
  }
  __syncthreads();

  // O = P * V : 5x4 tiles, K-dim 96 (padded) = 3 steps
  for (int t = wave; t < 20; t += 4) {
    int rt = t >> 2, ctd = t & 3;
    v8f o = (v8f){};
#pragma unroll
    for (int ks = 0; ks < 3; ++ks) {
      const bf16* pa = Ps + (rt * 16 + nn) * 104 + ks * 32;
      const bf16* pb = Vt + (ctd * 16 + nn) * 104 + ks * 32;
      v16bf a = frag16(pa + 16 * g, pa + 16 * g + 8);
      v16bf b = frag16(pb + 16 * g, pb + 16 * g + 8);
      o = wmma_bf16(a, b, o);
    }
#pragma unroll
    for (int r = 0; r < 8; ++r) {
      int row = rt * 16 + r + 8 * g;
      O[((size_t)win * LP + row) * CD + head * HD + ctd * 16 + nn] = (bf16)o[r];
    }
  }
}

// ---- host-side orchestration ------------------------------------------------

extern "C" void kernel_launch(void* const* d_in, const int* in_sizes, int n_in,
                              void* d_out, int out_size, void* d_ws, size_t ws_size,
                              hipStream_t stream) {
  const float* x   = (const float*)d_in[0];
  const float* y   = (const float*)d_in[1];
  const float* qc  = (const float*)d_in[4];
  const float* wq  = (const float*)d_in[5];
  const float* bq  = (const float*)d_in[6];
  const float* wk  = (const float*)d_in[7];
  const float* bk  = (const float*)d_in[8];
  const float* wv  = (const float*)d_in[9];
  const float* bv  = (const float*)d_in[10];
  const float* wo  = (const float*)d_in[11];
  const float* bo  = (const float*)d_in[12];
  const float* rpe = (const float*)d_in[13];
  float* out = (float*)d_out;

  char* ws = (char*)d_ws;
  const size_t SZ = (size_t)MTOT * CD * sizeof(bf16);   // 125,829,120 B
  bf16* xw  = (bf16*)ws;
  bf16* yw  = (bf16*)(ws + SZ);
  bf16* qb  = (bf16*)(ws + 2 * SZ);
  bf16* kb  = (bf16*)(ws + 3 * SZ);
  bf16* vb  = (bf16*)(ws + 4 * SZ);
  bf16* ao  = xw;                                       // reuse xw after Q GEMM
  bf16* wqb = (bf16*)(ws + 5 * SZ);
  bf16* wkb = wqb + (size_t)CD * CD;
  bf16* wvb = wkb + (size_t)CD * CD;
  bf16* wob = wvb + (size_t)CD * CD;

  const int nw = CD * CD;
  f32_to_bf16_kernel<<<(nw + 255) / 256, 256, 0, stream>>>(wq, wqb, nw);
  f32_to_bf16_kernel<<<(nw + 255) / 256, 256, 0, stream>>>(wk, wkb, nw);
  f32_to_bf16_kernel<<<(nw + 255) / 256, 256, 0, stream>>>(wv, wvb, nw);
  f32_to_bf16_kernel<<<(nw + 255) / 256, 256, 0, stream>>>(wo, wob, nw);

  pack_kernel<<<MTOT, 256, 0, stream>>>(x, y, qc, xw, yw);

  dim3 gg(MTOT / 128, CD / 128);   // (640, 6)
  wmma_gemm<<<gg, 256, 0, stream>>>(xw, wqb, bq, 0.125f, 0, qb, nullptr);
  wmma_gemm<<<gg, 256, 0, stream>>>(yw, wkb, bk, 1.0f,   0, kb, nullptr);
  wmma_gemm<<<gg, 256, 0, stream>>>(yw, wvb, bv, 1.0f,   0, vb, nullptr);

  attn_kernel<<<dim3(NWIN, HDS), 128, 0, stream>>>(qb, kb, vb, rpe, ao);

  wmma_gemm<<<gg, 256, 0, stream>>>(ao, wob, bo, 1.0f, 1, nullptr, out);
}
